// SpeechToTextModel_89275190215397
// MI455X (gfx1250) — matrix operand order repaired and checked by reading, not compiled, not run on hardware
//
#include <hip/hip_runtime.h>
#include <hip/hip_bf16.h>

typedef __attribute__((ext_vector_type(16))) _Float16 v16h;
typedef __attribute__((ext_vector_type(8)))  _Float16 v8h;
typedef __attribute__((ext_vector_type(8)))  float    v8f;

#define HH      128
#define G3H     384
#define BATCH   64
#define TSTEPS  4000
#define MROWS   (BATCH * TSTEPS)   // 256000, divisible by 128

// ---------------------------------------------------------------------------
// WMMA fragment loaders (f16, 16x16x32 shape), per ISA VGPR layout tables.
// ---------------------------------------------------------------------------
__device__ __forceinline__ v16h loadA16x32(const _Float16* tile, int ld, int l, int hi) {
  const _Float16* q = tile + (long)l * ld + hi * 8;
  v8h lo = *(const v8h*)(q);
  v8h hh = *(const v8h*)(q + 16);
  return __builtin_shufflevector(lo, hh, 0,1,2,3,4,5,6,7,8,9,10,11,12,13,14,15);
}
__device__ __forceinline__ v16h loadB32x16(const _Float16* tile, int ld, int l, int hi) {
  const _Float16* q = tile + (long)l * ld + hi * 8;
  v8h lo = *(const v8h*)(q);
  v8h hh = *(const v8h*)(q + 16);
  return __builtin_shufflevector(lo, hh, 0,1,2,3,4,5,6,7,8,9,10,11,12,13,14,15);
}

// Fast gate math on the scan critical path: v_exp2_f32 / v_rcp_f32 / v_tanh_f32
__device__ __forceinline__ float fast_sigmoid(float x) {
  float e = __builtin_amdgcn_exp2f(-1.442695041f * x);
  return __builtin_amdgcn_rcpf(1.f + e);
}
__device__ __forceinline__ float fast_tanh(float x) {
#if __has_builtin(__builtin_amdgcn_tanhf)
  return __builtin_amdgcn_tanhf(x);              // gfx1250 v_tanh_f32
#else
  float e = __builtin_amdgcn_exp2f(-2.885390082f * x);   // exp2(-2x*log2e)
  return 2.f * __builtin_amdgcn_rcpf(1.f + e) - 1.f;
#endif
}

// ---------------------------------------------------------------------------
// Prep: f32 -> f16 weight conversion (+K padding 26->32, N padding 136->144),
// and x -> f16 padded rows of 32.
// ---------------------------------------------------------------------------
__global__ __launch_bounds__(256)
void prep_kernel(const float* __restrict__ x,
                 const float* __restrict__ Wih0, const float* __restrict__ Whh0,
                 const float* __restrict__ Wih1, const float* __restrict__ Whh1,
                 const float* __restrict__ W1,   const float* __restrict__ W2,
                 _Float16* __restrict__ x16,  _Float16* __restrict__ wih0p,
                 _Float16* __restrict__ whh0, _Float16* __restrict__ wih1,
                 _Float16* __restrict__ whh1, _Float16* __restrict__ w1,
                 _Float16* __restrict__ w2p)
{
  long gt  = (long)blockIdx.x * blockDim.x + threadIdx.x;
  long nth = (long)gridDim.x * blockDim.x;
  for (long i = gt; i < (long)MROWS * 32; i += nth) {           // x [M,26] -> [M,32]
    long r = i >> 5; int c = (int)(i & 31);
    x16[i] = (c < 26) ? (_Float16)x[r * 26 + c] : (_Float16)0.f;
  }
  for (long i = gt; i < 384 * 32; i += nth) {                   // Wih0 [384,26]->[384,32]
    long r = i >> 5; int c = (int)(i & 31);
    wih0p[i] = (c < 26) ? (_Float16)Wih0[r * 26 + c] : (_Float16)0.f;
  }
  for (long i = gt; i < 384 * 128; i += nth) whh0[i] = (_Float16)Whh0[i];
  for (long i = gt; i < 384 * 128; i += nth) wih1[i] = (_Float16)Wih1[i];
  for (long i = gt; i < 384 * 128; i += nth) whh1[i] = (_Float16)Whh1[i];
  for (long i = gt; i < 128 * 128; i += nth) w1[i]   = (_Float16)W1[i];
  for (long i = gt; i < 144 * 128; i += nth) {                  // W2 [136,128]->[144,128]
    long r = i >> 7;
    w2p[i] = (r < 136) ? (_Float16)W2[i] : (_Float16)0.f;
  }
}

// ---------------------------------------------------------------------------
// Generic tall-skinny WMMA GEMM:  out[M,N] = A[M,K] * B[N,K]^T  (+bias, epilogue)
// OUTMODE 0: f32 + bias      1: f16 relu(·+bias)      2: f32 + bias, col<nmax
// ---------------------------------------------------------------------------
template<int KT, int NT, int OUTMODE>
__global__ __launch_bounds__(256)
void wmma_gemm_kernel(const _Float16* __restrict__ A, int lda,
                      const _Float16* __restrict__ B, int ldb,
                      const float* __restrict__ bias,
                      void* __restrict__ outv, int ldo, int nmax)
{
  const int wv = threadIdx.x >> 5;
  const int lane = threadIdx.x & 31;
  const int l = lane & 15, hi = lane >> 4;
  const long m0 = ((long)blockIdx.x * 8 + wv) * 16;

  v16h a[KT];
  const _Float16* Arow = A + m0 * lda;
#pragma unroll
  for (int k = 0; k < KT; ++k) a[k] = loadA16x32(Arow + k * 32, lda, l, hi);

  for (int nt = 0; nt < NT; ++nt) {
    v8f acc = {};
    const _Float16* Brow = B + (long)nt * 16 * ldb;
#pragma unroll
    for (int k = 0; k < KT; ++k) {
      v16h b = loadB32x16(Brow + k * 32, ldb, l, hi);
      acc = __builtin_amdgcn_wmma_f32_16x16x32_f16(false, a[k], false, b,
                                                   (short)0, acc, false, false);
    }
    const int col = nt * 16 + l;
    if (OUTMODE == 0) {
      float* out = (float*)outv;
      const float bc = bias[col];
#pragma unroll
      for (int v = 0; v < 8; ++v)
        out[(m0 + hi * 8 + v) * ldo + col] = acc[v] + bc;
    } else if (OUTMODE == 1) {
      _Float16* out = (_Float16*)outv;
      const float bc = bias[col];
#pragma unroll
      for (int v = 0; v < 8; ++v) {
        float r = acc[v] + bc;
        out[(m0 + hi * 8 + v) * ldo + col] = (_Float16)(r > 0.f ? r : 0.f);
      }
    } else {
      if (col < nmax) {
        float* out = (float*)outv;
        const float bc = bias[col];
#pragma unroll
        for (int v = 0; v < 8; ++v)
          out[(m0 + hi * 8 + v) * ldo + col] = acc[v] + bc;
      }
    }
  }
}

// ---------------------------------------------------------------------------
// GRU recurrence scan: one block = 16 batch rows, full T loop.
// Per step: gh[16,384] = h[16,128] @ Whh^T via 96 WMMAs (Whh preloaded into
// 96 VGPRs/lane of register B-fragments), then fused gate math in f32.
// gx loads issued at iteration top so their latency hides behind the WMMA
// block + barrier; bhh bias held in registers; h state in registers.
// ---------------------------------------------------------------------------
__global__ __launch_bounds__(256, 1)
void gru_scan_kernel(const float* __restrict__ gx,
                     const _Float16* __restrict__ Whh16,  // [384,128] f16 (N x K)
                     const float* __restrict__ bhh,
                     _Float16* __restrict__ hout)         // [M,128] f16
{
  __shared__ __align__(16) _Float16 h16[16 * HH];  // h as WMMA A source
  __shared__ __align__(16) float gh[16 * G3H];     // h@Whh^T result

  const int tid = threadIdx.x;
  const int wv = tid >> 5, lane = tid & 31;
  const int l = lane & 15, hi = lane >> 4;
  const int batch0 = blockIdx.x * 16;

  for (int i = tid; i < 16 * HH; i += 256) h16[i] = (_Float16)0.f;

  // Preload this wave's 48 columns of Whh as 12 register B-fragments (96 VGPRs).
  v16h Bt[3][4];
#pragma unroll
  for (int j = 0; j < 3; ++j)
#pragma unroll
    for (int k = 0; k < 4; ++k)
      Bt[j][k] = loadB32x16(Whh16 + (long)((wv * 3 + j) * 16) * HH + k * 32, HH, l, hi);

  // Elementwise ownership: thread owns batch row eb, hidden units [j0, j0+8)
  const int eb = tid >> 4;               // 0..15
  const int j0 = (tid & 15) * 8;         // 0,8,...,120

  // Loop-invariant bias in registers
  float br[8], bz[8], bn[8];
  {
    float4 a0 = *(const float4*)(bhh + j0);
    float4 a1 = *(const float4*)(bhh + j0 + 4);
    float4 b0 = *(const float4*)(bhh + HH + j0);
    float4 b1 = *(const float4*)(bhh + HH + j0 + 4);
    float4 c0 = *(const float4*)(bhh + 2 * HH + j0);
    float4 c1 = *(const float4*)(bhh + 2 * HH + j0 + 4);
    br[0]=a0.x; br[1]=a0.y; br[2]=a0.z; br[3]=a0.w; br[4]=a1.x; br[5]=a1.y; br[6]=a1.z; br[7]=a1.w;
    bz[0]=b0.x; bz[1]=b0.y; bz[2]=b0.z; bz[3]=b0.w; bz[4]=b1.x; bz[5]=b1.y; bz[6]=b1.z; bz[7]=b1.w;
    bn[0]=c0.x; bn[1]=c0.y; bn[2]=c0.z; bn[3]=c0.w; bn[4]=c1.x; bn[5]=c1.y; bn[6]=c1.z; bn[7]=c1.w;
  }

  float hreg[8];
#pragma unroll
  for (int v = 0; v < 8; ++v) hreg[v] = 0.f;

  // Running pointers (advance per timestep)
  const float* gxp = gx + ((long)(batch0 + eb) * TSTEPS) * G3H + j0;
  _Float16*  houtp = hout + ((long)(batch0 + eb) * TSTEPS) * HH + j0;
  const float* ghp = gh + eb * G3H + j0;
  _Float16* h16p = h16 + eb * HH + j0;

  __syncthreads();

  for (int t = 0; t < TSTEPS; ++t) {
    // --- Phase 1: issue gx loads early; latency hides behind WMMAs/barrier ---
    float4 xr0 = *(const float4*)(gxp);
    float4 xr1 = *(const float4*)(gxp + 4);
    float4 xz0 = *(const float4*)(gxp + HH);
    float4 xz1 = *(const float4*)(gxp + HH + 4);
    float4 xn0 = *(const float4*)(gxp + 2 * HH);
    float4 xn1 = *(const float4*)(gxp + 2 * HH + 4);

    // --- Phase 2: gh = h @ Whh^T  (A from LDS, B from registers) ---
    v16h a[4];
#pragma unroll
    for (int k = 0; k < 4; ++k) a[k] = loadA16x32(h16 + k * 32, HH, l, hi);
#pragma unroll
    for (int j = 0; j < 3; ++j) {
      v8f acc = {};
#pragma unroll
      for (int k = 0; k < 4; ++k)
        acc = __builtin_amdgcn_wmma_f32_16x16x32_f16(false, a[k], false, Bt[j][k],
                                                     (short)0, acc, false, false);
      const int n0 = (wv * 3 + j) * 16;
#pragma unroll
      for (int v = 0; v < 8; ++v)
        gh[(hi * 8 + v) * G3H + n0 + l] = acc[v];
    }
    __syncthreads();

    // --- Phase 3: fused gates ---
    {
      float4 hr0 = *(const float4*)(ghp);
      float4 hr1 = *(const float4*)(ghp + 4);
      float4 hz0 = *(const float4*)(ghp + HH);
      float4 hz1 = *(const float4*)(ghp + HH + 4);
      float4 hn0 = *(const float4*)(ghp + 2 * HH);
      float4 hn1 = *(const float4*)(ghp + 2 * HH + 4);
      float xr[8] = {xr0.x,xr0.y,xr0.z,xr0.w, xr1.x,xr1.y,xr1.z,xr1.w};
      float xz[8] = {xz0.x,xz0.y,xz0.z,xz0.w, xz1.x,xz1.y,xz1.z,xz1.w};
      float xn[8] = {xn0.x,xn0.y,xn0.z,xn0.w, xn1.x,xn1.y,xn1.z,xn1.w};
      float hr[8] = {hr0.x,hr0.y,hr0.z,hr0.w, hr1.x,hr1.y,hr1.z,hr1.w};
      float hz[8] = {hz0.x,hz0.y,hz0.z,hz0.w, hz1.x,hz1.y,hz1.z,hz1.w};
      float hn[8] = {hn0.x,hn0.y,hn0.z,hn0.w, hn1.x,hn1.y,hn1.z,hn1.w};
      v8h hnew16;
#pragma unroll
      for (int v = 0; v < 8; ++v) {
        float r = fast_sigmoid(xr[v] + hr[v] + br[v]);
        float z = fast_sigmoid(xz[v] + hz[v] + bz[v]);
        float n = fast_tanh(xn[v] + r * (hn[v] + bn[v]));
        float hnew = (1.f - z) * n + z * hreg[v];
        hreg[v] = hnew;
        hnew16[v] = (_Float16)hnew;
      }
      *(v8h*)h16p = hnew16;     // ds_store_b128
      *(v8h*)houtp = hnew16;    // global_store_b128
    }
    // Prefetch gx rows for t+8 (global_prefetch_b8)
    if (t + 8 < TSTEPS) __builtin_prefetch(gxp + 8 * G3H, 0, 3);

    gxp += G3H;
    houtp += HH;
    __syncthreads();
  }
}

// ---------------------------------------------------------------------------
extern "C" void kernel_launch(void* const* d_in, const int* in_sizes, int n_in,
                              void* d_out, int out_size, void* d_ws, size_t ws_size,
                              hipStream_t stream) {
  const float* x    = (const float*)d_in[0];
  const float* Wih0 = (const float*)d_in[1];
  const float* Whh0 = (const float*)d_in[2];
  const float* bih0 = (const float*)d_in[3];
  const float* bhh0 = (const float*)d_in[4];
  const float* Wih1 = (const float*)d_in[5];
  const float* Whh1 = (const float*)d_in[6];
  const float* bih1 = (const float*)d_in[7];
  const float* bhh1 = (const float*)d_in[8];
  const float* W1   = (const float*)d_in[9];
  const float* b1   = (const float*)d_in[10];
  const float* W2   = (const float*)d_in[11];
  const float* b2   = (const float*)d_in[12];
  float* out = (float*)d_out;

  // Workspace carve-out (~543 MB total)
  char* ws = (char*)d_ws;
  auto carve = [&](size_t bytes) {
    char* p = ws; ws += (bytes + 255) & ~(size_t)255; return p;
  };
  float*    gx    = (float*)   carve((size_t)MROWS * G3H * 4);   // 393 MB (reused both layers)
  _Float16* h1    = (_Float16*)carve((size_t)MROWS * HH * 2);    // 65.5 MB
  _Float16* h2    = (_Float16*)carve((size_t)MROWS * HH * 2);    // 65.5 MB
  _Float16* x16   = (_Float16*)carve((size_t)MROWS * 32 * 2);    // 16.4 MB
  _Float16* wih0p = (_Float16*)carve(384 * 32 * 2);
  _Float16* whh0f = (_Float16*)carve(384 * 128 * 2);
  _Float16* wih1f = (_Float16*)carve(384 * 128 * 2);
  _Float16* whh1f = (_Float16*)carve(384 * 128 * 2);
  _Float16* w1f   = (_Float16*)carve(128 * 128 * 2);
  _Float16* w2p   = (_Float16*)carve(144 * 128 * 2);
  _Float16* hidden = (_Float16*)gx;  // gx dead after second scan; alias for ReLU output

  prep_kernel<<<2048, 256, 0, stream>>>(x, Wih0, Whh0, Wih1, Whh1, W1, W2,
                                        x16, wih0p, whh0f, wih1f, whh1f, w1f, w2p);

  const int gblk = MROWS / 128;  // 2000 blocks, 8 m-tiles each

  // Layer 0: gx = x @ Wih0^T + bih0  (K padded 26->32)
  wmma_gemm_kernel<1, 24, 0><<<gblk, 256, 0, stream>>>(x16, 32, wih0p, 32, bih0, gx, G3H, G3H);
  gru_scan_kernel<<<BATCH / 16, 256, 0, stream>>>(gx, whh0f, bhh0, h1);

  // Layer 1: gx = h1 @ Wih1^T + bih1
  wmma_gemm_kernel<4, 24, 0><<<gblk, 256, 0, stream>>>(h1, 128, wih1f, 128, bih1, gx, G3H, G3H);
  gru_scan_kernel<<<BATCH / 16, 256, 0, stream>>>(gx, whh1f, bhh1, h2);

  // Head: hidden = relu(h2 @ W1^T + b1);  out = hidden @ W2^T + b2  (N=136, padded 144)
  wmma_gemm_kernel<4, 8, 1><<<gblk, 256, 0, stream>>>(h2, 128, w1f, 128, b1, hidden, 128, 128);
  wmma_gemm_kernel<4, 9, 2><<<gblk, 256, 0, stream>>>(hidden, 128, w2p, 128, b2, out, 136, 136);
}
